// MyPointConv_2860448219714
// MI455X (gfx1250) — matrix operand order repaired and checked by reading, not compiled, not run on hardware
//
#include <hip/hip_runtime.h>
#include <hip/hip_fp16.h>
#include <stdint.h>

typedef __attribute__((ext_vector_type(16))) _Float16 v16h;
typedef __attribute__((ext_vector_type(8)))  _Float16 v8h;
typedef __attribute__((ext_vector_type(4)))  _Float16 v4h;
typedef __attribute__((ext_vector_type(8)))  float    v8f;

#define N_NODES 50000
#define N_EDGES 800000
#define D_FEAT  128
#define M_TOT   (N_EDGES + N_NODES)       /* 850000 messages (incl. self loops) */
#define NGROUPS1 (M_TOT / 16)             /* 53125, exact */
#define NPAIRS1  ((NGROUPS1 + 1) / 2)     /* 26563 wave-pairs (last is half) */
#define NGROUPS2 (N_NODES / 16)           /* 3125, exact */

#define AGG_WORDS ((size_t)N_NODES * D_FEAT)        /* 6.4M uints */
#define AGG_BYTES (AGG_WORDS * 4)                   /* 25,600,000 B (32B aligned) */
#define W1F_HALVES (5 * 8 * 32 * 16)                /* K padded 131->160: 5 chunks */
#define W2F_HALVES (4 * 8 * 32 * 16)                /* K = 128: 4 chunks */
#define XH_OFFSET (AGG_BYTES + (W1F_HALVES + W2F_HALVES) * 2)   /* 16B aligned */

// ---------------------------------------------------------------------------
// agg init: 0.0f == uint 0; ReLU output >= 0 so uint-bit max == float max.
__global__ void zero_agg_kernel(uint4* __restrict__ agg4) {
    size_t i = (size_t)blockIdx.x * blockDim.x + threadIdx.x;
    if (i < AGG_WORDS / 4) agg4[i] = make_uint4(0u, 0u, 0u, 0u);
}

// x (fp32) -> xh (f16), halves gather traffic and enables async-to-LDS staging.
__global__ void cvt_x_kernel(const float4* __restrict__ x4, v4h* __restrict__ xh4) {
    size_t i = (size_t)blockIdx.x * blockDim.x + threadIdx.x;
    if (i < (size_t)N_NODES * 32) {
        float4 v = x4[i];
        v4h h = { (_Float16)v.x, (_Float16)v.y, (_Float16)v.z, (_Float16)v.w };
        xh4[i] = h;
    }
}

// ---------------------------------------------------------------------------
// Repack W [K x 128] row-major fp32 into WMMA B fragments (f16).
// Fragment (c,t): lane l holds column N = t*16 + (l&15), K = c*32 + (h>>3)*16 + (l>>4)*8 + (h&7).
__global__ void repack_w1_kernel(const float* __restrict__ W1, _Float16* __restrict__ out) {
    int idx = blockIdx.x * 256 + threadIdx.x;
    if (idx >= W1F_HALVES) return;
    int h = idx & 15, l = (idx >> 4) & 31, t = (idx >> 9) & 7, c = idx >> 12;
    int n = t * 16 + (l & 15);
    int k = c * 32 + (h >> 3) * 16 + (l >> 4) * 8 + (h & 7);
    float v = (k < 131) ? W1[k * 128 + n] : 0.0f;   // zero-pad K 131..159
    out[idx] = (_Float16)v;
}

__global__ void repack_w2_kernel(const float* __restrict__ W2, _Float16* __restrict__ out) {
    int idx = blockIdx.x * 256 + threadIdx.x;
    if (idx >= W2F_HALVES) return;
    int h = idx & 15, l = (idx >> 4) & 31, t = (idx >> 9) & 7, c = idx >> 12;
    int n = t * 16 + (l & 15);
    int k = c * 32 + (h >> 3) * 16 + (l >> 4) * 8 + (h & 7);             // always < 128
    out[idx] = (_Float16)W2[k * 128 + n];
}

// ---------------------------------------------------------------------------
// Fused edge kernel, M-blocked by 2: each wave owns 32 edges (two 16-row tiles),
// stages A via global_load_async_to_lds_b128, reuses every B fragment twice,
// then ReLU + monotonic-filtered atomicMax scatter into agg[dst].
__global__ __launch_bounds__(256)
void edge_gemm_max_kernel(const _Float16* __restrict__ xh,
                          const float* __restrict__ pos,
                          const int*   __restrict__ ei,
                          const _Float16* __restrict__ w1f,
                          const float* __restrict__ b1,
                          unsigned int* __restrict__ agg) {
    // 32 rows x 160 halves (K padded), stride 168 halves = 336 B (16B aligned).
    __shared__ __align__(16) _Float16 lds[8][32][168];
    __shared__ int dsti[8][32];

    const int lane = threadIdx.x & 31;
    const int w    = threadIdx.x >> 5;
    const int gp   = blockIdx.x * 8 + w;          // wave-pair index
    const bool active = (gp < NPAIRS1);
    const int ebase = gp * 32;

    if (active) {
        // --- Async gather: 32 rows x 256 B = 512 b128 transfers (16 per lane).
        // row = i*2 + (lane>>4), chunk = lane&15  (16 B per chunk).
        #pragma unroll
        for (int i = 0; i < 16; ++i) {
            int row   = i * 2 + (lane >> 4);
            int chunk = lane & 15;
            int e  = ebase + row;
            int ec = (e < M_TOT) ? e : (M_TOT - 1);          // clamp tail rows
            int s  = (ec < N_EDGES) ? ei[ec] : (ec - N_EDGES);
            const _Float16* gsrc = xh + (size_t)s * 128 + chunk * 8;
            unsigned lofs = (unsigned)(uintptr_t)&lds[w][row][chunk * 8];
            asm volatile("global_load_async_to_lds_b128 %0, %1, off"
                         :: "v"(lofs), "v"(gsrc) : "memory");
        }
        // --- Tail columns: dpos (128..130) + zero pad (131..159); 1 row per lane.
        {
            int row = lane;
            int e  = ebase + row;
            int ec = (e < M_TOT) ? e : (M_TOT - 1);
            int s, dn;
            if (ec < N_EDGES) { s = ei[ec]; dn = ei[N_EDGES + ec]; }
            else              { s = dn = ec - N_EDGES; }       // self loop: dpos = 0
            dsti[w][row] = dn;
            lds[w][row][128] = (_Float16)(pos[s * 3 + 0] - pos[dn * 3 + 0]);
            lds[w][row][129] = (_Float16)(pos[s * 3 + 1] - pos[dn * 3 + 1]);
            lds[w][row][130] = (_Float16)(pos[s * 3 + 2] - pos[dn * 3 + 2]);
            #pragma unroll
            for (int k = 131; k < 160; ++k) lds[w][row][k] = (_Float16)0.0f;
        }
    }
    asm volatile("s_wait_asynccnt 0" ::: "memory");   // async LDS writes visible
    __syncthreads();
    if (!active) return;

    v8f acc[2][8] = {};
    const _Float16* row0 = &lds[w][(lane & 15)][0];
    const _Float16* row1 = &lds[w][16 + (lane & 15)][0];
    const int koff = (lane >> 4) * 8;   // A layout: lanes>=16 hold K-offset +8 / +24

    #pragma unroll
    for (int c = 0; c < 5; ++c) {
        v8h a0l = *(const v8h*)(row0 + c * 32 + koff);
        v8h a0h = *(const v8h*)(row0 + c * 32 + 16 + koff);
        v8h a1l = *(const v8h*)(row1 + c * 32 + koff);
        v8h a1h = *(const v8h*)(row1 + c * 32 + 16 + koff);
        v16h a0 = __builtin_shufflevector(a0l, a0h, 0,1,2,3,4,5,6,7,8,9,10,11,12,13,14,15);
        v16h a1 = __builtin_shufflevector(a1l, a1h, 0,1,2,3,4,5,6,7,8,9,10,11,12,13,14,15);
        #pragma unroll
        for (int t = 0; t < 8; ++t) {
            v16h b = *(const v16h*)(w1f + ((c * 8 + t) * 32 + lane) * 16);
            acc[0][t] = __builtin_amdgcn_wmma_f32_16x16x32_f16(
                false, a0, false, b, (short)0, acc[0][t], false, false);
            acc[1][t] = __builtin_amdgcn_wmma_f32_16x16x32_f16(
                false, a1, false, b, (short)0, acc[1][t], false, false);
        }
    }

    // Epilogue: C/D layout -> M = v + (lane>>4)*8, N = t*16 + (lane&15).
    // agg grows monotonically from 0: skip RMW unless candidate can win.
    const int mrow = (lane >> 4) * 8;
    #pragma unroll
    for (int q = 0; q < 2; ++q) {
        #pragma unroll
        for (int t = 0; t < 8; ++t) {
            int n = t * 16 + (lane & 15);
            float bias = b1[n];
            #pragma unroll
            for (int v = 0; v < 8; ++v) {
                int m = q * 16 + mrow + v;
                int e = ebase + m;
                if (e < M_TOT) {
                    float val = acc[q][t][v] + bias;
                    val = val > 0.0f ? val : 0.0f;            // ReLU => val >= 0
                    unsigned bits = __float_as_uint(val);
                    size_t idx = (size_t)dsti[w][m] * D_FEAT + n;
                    unsigned cur = agg[idx];                   // monotone filter
                    if (bits > cur) atomicMax(&agg[idx], bits);
                }
            }
        }
    }
}

// ---------------------------------------------------------------------------
// Output GEMM: out = agg @ W2 + b2
__global__ __launch_bounds__(256)
void out_gemm_kernel(const unsigned int* __restrict__ agg,
                     const _Float16* __restrict__ w2f,
                     const float* __restrict__ b2,
                     float* __restrict__ out) {
    __shared__ __align__(16) _Float16 lds[8][16][136];  // stride 272 B (16B aligned)
    const int lane = threadIdx.x & 31;
    const int w    = threadIdx.x >> 5;
    const int g    = blockIdx.x * 8 + w;
    const bool active = (g < NGROUPS2);

    if (active) {
        const uint4* a4 = (const uint4*)agg;
        #pragma unroll 4
        for (int r = 0; r < 16; ++r) {
            uint4 u = a4[(size_t)(g * 16 + r) * 32 + lane];
            v4h hv = { (_Float16)__uint_as_float(u.x), (_Float16)__uint_as_float(u.y),
                       (_Float16)__uint_as_float(u.z), (_Float16)__uint_as_float(u.w) };
            *(v4h*)&lds[w][r][lane * 4] = hv;
        }
    }
    __syncthreads();
    if (!active) return;

    v8f acc[8] = {};
    const _Float16* rowbase = &lds[w][lane & 15][0];
    const int koff = (lane >> 4) * 8;

    #pragma unroll
    for (int c = 0; c < 4; ++c) {
        v8h alo = *(const v8h*)(rowbase + c * 32 + koff);
        v8h ahi = *(const v8h*)(rowbase + c * 32 + 16 + koff);
        v16h a = __builtin_shufflevector(alo, ahi, 0,1,2,3,4,5,6,7,8,9,10,11,12,13,14,15);
        #pragma unroll
        for (int t = 0; t < 8; ++t) {
            v16h b = *(const v16h*)(w2f + ((c * 8 + t) * 32 + lane) * 16);
            acc[t] = __builtin_amdgcn_wmma_f32_16x16x32_f16(
                false, a, false, b, (short)0, acc[t], false, false);
        }
    }

    const int mrow = (lane >> 4) * 8;
    #pragma unroll
    for (int t = 0; t < 8; ++t) {
        int n = t * 16 + (lane & 15);
        float bias = b2[n];
        #pragma unroll
        for (int v = 0; v < 8; ++v) {
            int m = g * 16 + mrow + v;
            out[(size_t)m * D_FEAT + n] = acc[t][v] + bias;
        }
    }
}

// ---------------------------------------------------------------------------
extern "C" void kernel_launch(void* const* d_in, const int* in_sizes, int n_in,
                              void* d_out, int out_size, void* d_ws, size_t ws_size,
                              hipStream_t stream) {
    const float* x   = (const float*)d_in[0];
    const float* pos = (const float*)d_in[1];
    const int*   ei  = (const int*)d_in[2];
    const float* W1  = (const float*)d_in[3];
    const float* b1  = (const float*)d_in[4];
    const float* W2  = (const float*)d_in[5];
    const float* b2  = (const float*)d_in[6];
    float* out = (float*)d_out;

    unsigned int* agg = (unsigned int*)d_ws;
    _Float16* w1f = (_Float16*)((char*)d_ws + AGG_BYTES);
    _Float16* w2f = w1f + W1F_HALVES;
    _Float16* xh  = (_Float16*)((char*)d_ws + XH_OFFSET);

    // 1) agg = 0 (valid init for max of non-negative ReLU outputs)
    {
        int n4 = (int)(AGG_WORDS / 4);
        zero_agg_kernel<<<(n4 + 255) / 256, 256, 0, stream>>>((uint4*)agg);
    }
    // 2) x -> f16 copy; repack weights into WMMA B-fragment layout
    {
        int n4 = N_NODES * 32;
        cvt_x_kernel<<<(n4 + 255) / 256, 256, 0, stream>>>((const float4*)x, (v4h*)xh);
    }
    repack_w1_kernel<<<(W1F_HALVES + 255) / 256, 256, 0, stream>>>(W1, w1f);
    repack_w2_kernel<<<(W2F_HALVES + 255) / 256, 256, 0, stream>>>(W2, w2f);
    // 3) fused message GEMM + ReLU + scatter-max (8 waves/block, 32 edges/wave)
    edge_gemm_max_kernel<<<(NPAIRS1 + 7) / 8, 256, 0, stream>>>(xh, pos, ei, w1f, b1, agg);
    // 4) out = agg @ W2 + b2
    out_gemm_kernel<<<(NGROUPS2 + 7) / 8, 256, 0, stream>>>(agg, w2f, b2, out);
}
